// DiscretizedDPLRSSMBlock_9680856285215
// MI455X (gfx1250) — matrix operand order repaired and checked by reading, not compiled, not run on hardware
//
#include <hip/hip_runtime.h>

#define HIDDEN 4096
#define BATCHSZ 4096
#define DELTA 1.0f
#define KBLK 32
#define LSTR 40   // padded LDS row stride in bf16 elems (80 B: 16B-aligned rows, bank-spread)

typedef __attribute__((ext_vector_type(2)))  float  v2f;
typedef __attribute__((ext_vector_type(4)))  float  v4f;
typedef __attribute__((ext_vector_type(8)))  float  v8f;
typedef __attribute__((ext_vector_type(2)))  __bf16 v2bf;
typedef __attribute__((ext_vector_type(4)))  __bf16 v4bf;
typedef __attribute__((ext_vector_type(8)))  __bf16 v8bf;
typedef __attribute__((ext_vector_type(16))) __bf16 v16bf;

#define SHUF16(a, b) __builtin_shufflevector(a, b, 0,1,2,3,4,5,6,7,8,9,10,11,12,13,14,15)

// Async 16B copy global -> LDS (CDNA5 GLOBAL_LOAD_ASYNC_TO_LDS_B128, GVS mode, ASYNCcnt).
#define ASYNC_CP16(ldsoff_u32, goff_u32, sbase_ptr)                          \
    asm volatile("global_load_async_to_lds_b128 %0, %1, %2"                  \
                 :: "v"(ldsoff_u32), "v"(goff_u32), "s"(sbase_ptr) : "memory")

static __device__ __forceinline__ void wait_async0() {
    asm volatile("s_wait_asynccnt 0x0" ::: "memory");
}

static __device__ __forceinline__ unsigned lds_off(const void* p) {
    // generic shared-aperture address keeps the wave-relative LDS byte offset in [31:0]
    return (unsigned)(unsigned long long)p;
}

// ---- split-precision helper: v = (float)hi + (float)lo, hi/lo bf16 (RNE) ----
__device__ __forceinline__ void split_bf16(float v, __bf16& hi, __bf16& lo) {
    hi = (__bf16)v;
    lo = (__bf16)(v - (float)hi);
}

// ---------------- Kernel 1: hq = h @ q_vec   -> [BATCH x 4] ----------------
__global__ __launch_bounds__(256) void hq_kernel(const float* __restrict__ h,
                                                 const float* __restrict__ q,
                                                 float* __restrict__ hq) {
    __shared__ __align__(16) v4f red[256];
    const int row = blockIdx.x;
    const int tid = threadIdx.x;
    const float* hrow = h + (size_t)row * HIDDEN;
    v4f s = {0.0f, 0.0f, 0.0f, 0.0f};
    for (int k = tid; k < HIDDEN; k += 256) {
        const float hv = __builtin_nontemporal_load(hrow + k);
        const v4f qv = *(const v4f*)(q + 4 * k);
        s += hv * qv;
    }
    red[tid] = s;
    __syncthreads();
    for (int off = 128; off > 0; off >>= 1) {
        if (tid < off) red[tid] += red[tid + off];
        __syncthreads();
    }
    if (tid == 0) *(v4f*)(hq + 4 * row) = red[0];
}

// ---------------- Kernel 2a: elementwise split  src(f32) -> hi,lo (bf16) ----------------
__global__ __launch_bounds__(256) void convert_split_kernel(const float* __restrict__ src,
                                                            __bf16* __restrict__ hi,
                                                            __bf16* __restrict__ lo) {
    const size_t base = ((size_t)blockIdx.x * 256 + threadIdx.x) * 4;
    const v4f v = *(const v4f*)(src + base);
    v4bf h4, l4;
#pragma unroll
    for (int e = 0; e < 4; ++e) { __bf16 hb, lb; split_bf16(v[e], hb, lb); h4[e] = hb; l4[e] = lb; }
    *(v4bf*)(hi + base) = h4;
    *(v4bf*)(lo + base) = l4;
}

// ------- Kernel 2b: split + transpose  b_mat[k][n](f32) -> bhiT,bloT[n][k](bf16) -------
#define TS 72
__global__ __launch_bounds__(256) void convert_bT_kernel(const float* __restrict__ b,
                                                         __bf16* __restrict__ bhiT,
                                                         __bf16* __restrict__ bloT) {
    __shared__ __align__(16) __bf16 Thi[64 * TS];
    __shared__ __align__(16) __bf16 Tlo[64 * TS];
    const int tid = threadIdx.x;
    const int k0 = blockIdx.y * 64;
    const int n0 = blockIdx.x * 64;
#pragma unroll
    for (int p = 0; p < 4; ++p) {
        const int krow = p * 16 + (tid >> 4);
        const int nc4  = (tid & 15) << 2;
        const v4f v = *(const v4f*)(b + (size_t)(k0 + krow) * HIDDEN + n0 + nc4);
#pragma unroll
        for (int e = 0; e < 4; ++e) {
            __bf16 hb, lb; split_bf16(v[e], hb, lb);
            Thi[(nc4 + e) * TS + krow] = hb;
            Tlo[(nc4 + e) * TS + krow] = lb;
        }
    }
    __syncthreads();
#pragma unroll
    for (int i = 0; i < 2; ++i) {
        const int ch = tid + i * 256;      // 0..511 chunks of 8 bf16
        const int n  = ch >> 3;
        const int c8 = (ch & 7) << 3;
        const v8bf hv = *(const v8bf*)(&Thi[n * TS + c8]);
        const v8bf lv = *(const v8bf*)(&Tlo[n * TS + c8]);
        *(v8bf*)(bhiT + (size_t)(n0 + n) * HIDDEN + k0 + c8) = hv;
        *(v8bf*)(bloT + (size_t)(n0 + n) * HIDDEN + k0 + c8) = lv;
    }
}

// ================= GEMM core =================
// WG: 256 threads = 8 waves (2x4). WG tile 128x256, wave tile 64x64 (4x4 subtiles).
// bf16x3 split GEMM: x*b ~= xhi*bhi + xhi*blo + xlo*bhi, f32 WMMA accumulation.
struct Frags { v16bf ahi[4], alo[4], bhi[4], blo[4]; };

__device__ __forceinline__ void load_frags(
    const __bf16* Xhi, const __bf16* Xlo, const __bf16* Bhi, const __bf16* Blo,
    int wm, int wn, int lane15, int laneHi, Frags& f)
{
    const __bf16* xph = &Xhi[(wm * 64 + lane15) * LSTR + 8 * laneHi];
    const __bf16* xpl = &Xlo[(wm * 64 + lane15) * LSTR + 8 * laneHi];
    const __bf16* bph = &Bhi[(wn * 64 + lane15) * LSTR + 8 * laneHi];
    const __bf16* bpl = &Blo[(wn * 64 + lane15) * LSTR + 8 * laneHi];
#pragma unroll
    for (int mi = 0; mi < 4; ++mi) {
        const int o = mi * 16 * LSTR;
        v8bf h0 = *(const v8bf*)(xph + o);
        v8bf h1 = *(const v8bf*)(xph + o + 16);
        f.ahi[mi] = SHUF16(h0, h1);
        v8bf l0 = *(const v8bf*)(xpl + o);
        v8bf l1 = *(const v8bf*)(xpl + o + 16);
        f.alo[mi] = SHUF16(l0, l1);
    }
#pragma unroll
    for (int nj = 0; nj < 4; ++nj) {
        const int o = nj * 16 * LSTR;
        v8bf bh0 = *(const v8bf*)(bph + o);
        v8bf bh1 = *(const v8bf*)(bph + o + 16);
        f.bhi[nj] = SHUF16(bh0, bh1);
        v8bf bl0 = *(const v8bf*)(bpl + o);
        v8bf bl1 = *(const v8bf*)(bpl + o + 16);
        f.blo[nj] = SHUF16(bl0, bl1);
    }
}

__device__ __forceinline__ void compute_48(const Frags& f, v8f (&acc)[4][4])
{
#pragma unroll
    for (int nj = 0; nj < 4; ++nj)
#pragma unroll
        for (int mi = 0; mi < 4; ++mi) {
            acc[mi][nj] = __builtin_amdgcn_wmma_f32_16x16x32_bf16(
                false, f.ahi[mi], false, f.bhi[nj], (short)0, acc[mi][nj], false, false);
            acc[mi][nj] = __builtin_amdgcn_wmma_f32_16x16x32_bf16(
                false, f.ahi[mi], false, f.blo[nj], (short)0, acc[mi][nj], false, false);
            acc[mi][nj] = __builtin_amdgcn_wmma_f32_16x16x32_bf16(
                false, f.alo[mi], false, f.bhi[nj], (short)0, acc[mi][nj], false, false);
        }
}

__device__ __forceinline__ void epilogue_store(
    const float* __restrict__ h, const float* __restrict__ a_diag,
    const float* __restrict__ p_vec, const float* __restrict__ hq,
    float* __restrict__ out, int m_wave, int n_wave,
    int lane15, int laneHi, v8f (&acc)[4][4])
{
#pragma unroll
    for (int nj = 0; nj < 4; ++nj) {
        const int col = n_wave + nj * 16 + lane15;
        const float ac = 1.0f + DELTA * a_diag[col];
        const v4f pc = *(const v4f*)(p_vec + 4 * col);
#pragma unroll
        for (int mi = 0; mi < 4; ++mi) {
#pragma unroll
            for (int v = 0; v < 8; ++v) {
                const int row = m_wave + mi * 16 + v + 8 * laneHi;
                const v4f hqv = *(const v4f*)(hq + 4 * row);
                const float lr = hqv[0] * pc[0] + hqv[1] * pc[1] +
                                 hqv[2] * pc[2] + hqv[3] * pc[3];
                const size_t idx = (size_t)row * HIDDEN + col;
                const float hv = __builtin_nontemporal_load(h + idx);
                const float res = DELTA * acc[mi][nj][v] + hv * ac + DELTA * lr;
                __builtin_nontemporal_store(res, out + idx);
            }
        }
    }
}

// ---------------- Kernel 3 (fast path): async-to-LDS staging, register-pipelined ----------------
__global__ __launch_bounds__(256) void gemm_pre_kernel(
    const float* __restrict__ h,
    const __bf16* __restrict__ xhiG, const __bf16* __restrict__ xloG,
    const float* __restrict__ a_diag, const float* __restrict__ p_vec,
    const __bf16* __restrict__ bhiT, const __bf16* __restrict__ bloT,
    const float* __restrict__ hq, float* __restrict__ out)
{
    __shared__ __align__(16) __bf16 Xhi[128 * LSTR];
    __shared__ __align__(16) __bf16 Xlo[128 * LSTR];
    __shared__ __align__(16) __bf16 Bhi[256 * LSTR];
    __shared__ __align__(16) __bf16 Blo[256 * LSTR];

    const int tid    = threadIdx.x;
    const int lane   = tid & 31;
    const int wave   = tid >> 5;
    const int lane15 = lane & 15;
    const int laneHi = lane >> 4;
    const int wm = wave >> 2;
    const int wn = wave & 3;
    const int m_base = blockIdx.y * 128;
    const int n_base = blockIdx.x * 256;

    // per-thread staging coordinates (constant across iterations)
    const int xr   = (tid + 0) >> 2;                 // handled below per chunk
    (void)xr;

    auto issue_async = [&](int k0) {
        // X tiles: 128 rows x 32 k bf16 = 512 chunks of 16B per buffer (2/thread)
#pragma unroll
        for (int i = 0; i < 2; ++i) {
            const int ch  = tid + i * 256;           // 0..511
            const int r   = ch >> 2;                 // row 0..127
            const int c16 = (ch & 3) << 3;           // elem 0,8,16,24
            const unsigned goff = (unsigned)(((m_base + r) * HIDDEN + k0 + c16) * 2);
            ASYNC_CP16(lds_off(&Xhi[r * LSTR + c16]), goff, xhiG);
            ASYNC_CP16(lds_off(&Xlo[r * LSTR + c16]), goff, xloG);
        }
        // B^T tiles: 256 rows x 32 k bf16 = 1024 chunks per buffer (4/thread)
#pragma unroll
        for (int i = 0; i < 4; ++i) {
            const int ch  = tid + i * 256;           // 0..1023
            const int n   = ch >> 2;                 // row 0..255
            const int c16 = (ch & 3) << 3;
            const unsigned goff = (unsigned)(((n_base + n) * HIDDEN + k0 + c16) * 2);
            ASYNC_CP16(lds_off(&Bhi[n * LSTR + c16]), goff, bhiT);
            ASYNC_CP16(lds_off(&Blo[n * LSTR + c16]), goff, bloT);
        }
    };

    v8f acc[4][4];
#pragma unroll
    for (int mi = 0; mi < 4; ++mi)
#pragma unroll
        for (int nj = 0; nj < 4; ++nj)
            acc[mi][nj] = (v8f){0.f, 0.f, 0.f, 0.f, 0.f, 0.f, 0.f, 0.f};

    issue_async(0);                                  // preamble: DMA first tile
    for (int k0 = 0; k0 < HIDDEN; k0 += KBLK) {
        wait_async0();                               // own share of tile k0 done
        __syncthreads();                             // everyone's share done
        Frags f;
        load_frags(Xhi, Xlo, Bhi, Blo, wm, wn, lane15, laneHi, f);
        __syncthreads();                             // ds reads drained -> LDS reusable
        if (k0 + KBLK < HIDDEN) issue_async(k0 + KBLK);  // DMA next tile under WMMA burst
        compute_48(f, acc);
    }
    epilogue_store(h, a_diag, p_vec, hq, out,
                   m_base + wm * 64, n_base + wn * 64, lane15, laneHi, acc);
}

// ---------------- Kernel 3 (fallback): self-converting (round-2 scheme) ----------------
__global__ __launch_bounds__(256) void gemm_fused_kernel(
    const float* __restrict__ h, const float* __restrict__ x,
    const float* __restrict__ a_diag, const float* __restrict__ p_vec,
    const float* __restrict__ b_mat, const float* __restrict__ hq,
    float* __restrict__ out)
{
    __shared__ __align__(16) __bf16 Xhi[128 * LSTR];
    __shared__ __align__(16) __bf16 Xlo[128 * LSTR];
    __shared__ __align__(16) __bf16 Bhi[256 * LSTR];
    __shared__ __align__(16) __bf16 Blo[256 * LSTR];

    const int tid    = threadIdx.x;
    const int lane   = tid & 31;
    const int wave   = tid >> 5;
    const int lane15 = lane & 15;
    const int laneHi = lane >> 4;
    const int wm = wave >> 2;
    const int wn = wave & 3;
    const int m_base = blockIdx.y * 128;
    const int n_base = blockIdx.x * 256;

    v8f acc[4][4];
#pragma unroll
    for (int mi = 0; mi < 4; ++mi)
#pragma unroll
        for (int nj = 0; nj < 4; ++nj)
            acc[mi][nj] = (v8f){0.f, 0.f, 0.f, 0.f, 0.f, 0.f, 0.f, 0.f};

    for (int k0 = 0; k0 < HIDDEN; k0 += KBLK) {
#pragma unroll
        for (int i = 0; i < 4; ++i) {
            const int f4 = tid + i * 256;
            const int r  = f4 >> 3;
            const int c4 = (f4 & 7) << 2;
            const v4f v = *(const v4f*)(x + (size_t)(m_base + r) * HIDDEN + k0 + c4);
            v4bf h4, l4;
#pragma unroll
            for (int e = 0; e < 4; ++e) { __bf16 hb, lb; split_bf16(v[e], hb, lb); h4[e] = hb; l4[e] = lb; }
            *(v4bf*)(&Xhi[r * LSTR + c4]) = h4;
            *(v4bf*)(&Xlo[r * LSTR + c4]) = l4;
        }
#pragma unroll
        for (int i = 0; i < 4; ++i) {
            const int pid = tid + i * 256;
            const int kp  = pid >> 6;
            const int c4  = (pid & 63) << 2;
            const float* bp = b_mat + (size_t)(k0 + 2 * kp) * HIDDEN + n_base + c4;
            const v4f r0 = *(const v4f*)(bp);
            const v4f r1 = *(const v4f*)(bp + HIDDEN);
#pragma unroll
            for (int e = 0; e < 4; ++e) {
                __bf16 h0, l0, h1, l1;
                split_bf16(r0[e], h0, l0);
                split_bf16(r1[e], h1, l1);
                v2bf hh = {h0, h1}, ll = {l0, l1};
                *(v2bf*)(&Bhi[(c4 + e) * LSTR + 2 * kp]) = hh;
                *(v2bf*)(&Blo[(c4 + e) * LSTR + 2 * kp]) = ll;
            }
        }
        __syncthreads();
        Frags f;
        load_frags(Xhi, Xlo, Bhi, Blo, wm, wn, lane15, laneHi, f);
        compute_48(f, acc);
        __syncthreads();
    }
    epilogue_store(h, a_diag, p_vec, hq, out,
                   m_base + wm * 64, n_base + wn * 64, lane15, laneHi, acc);
}

extern "C" void kernel_launch(void* const* d_in, const int* in_sizes, int n_in,
                              void* d_out, int out_size, void* d_ws, size_t ws_size,
                              hipStream_t stream) {
    const float* h      = (const float*)d_in[0];
    const float* x      = (const float*)d_in[1];
    const float* a_diag = (const float*)d_in[2];
    const float* p_vec  = (const float*)d_in[3];
    const float* q_vec  = (const float*)d_in[4];
    const float* b_mat  = (const float*)d_in[5];
    float* out = (float*)d_out;

    char* ws = (char*)d_ws;
    float*  hq   = (float*)ws;                                // 64 KB
    const size_t NELEM = (size_t)HIDDEN * HIDDEN;             // 16M
    __bf16* xhiG = (__bf16*)(ws + (1 << 20));                 // 32 MB each
    __bf16* xloG = xhiG + NELEM;
    __bf16* bhiT = xloG + NELEM;
    __bf16* bloT = bhiT + NELEM;
    const size_t needed = (1 << 20) + 4 * NELEM * sizeof(__bf16);

    hq_kernel<<<BATCHSZ, 256, 0, stream>>>(h, q_vec, hq);
    dim3 grid(HIDDEN / 256, BATCHSZ / 128);

    if (ws_size >= needed) {
        convert_split_kernel<<<(int)(NELEM / 1024), 256, 0, stream>>>(x, xhiG, xloG);
        convert_bT_kernel<<<dim3(HIDDEN / 64, HIDDEN / 64), 256, 0, stream>>>(b_mat, bhiT, bloT);
        gemm_pre_kernel<<<grid, 256, 0, stream>>>(h, xhiG, xloG, a_diag, p_vec,
                                                  bhiT, bloT, hq, out);
    } else {
        gemm_fused_kernel<<<grid, 256, 0, stream>>>(h, x, a_diag, p_vec, b_mat, hq, out);
    }
}
